// GPT_76725295776296
// MI455X (gfx1250) — matrix operand order repaired and checked by reading, not compiled, not run on hardware
//
#include <hip/hip_runtime.h>
#include <math.h>

// ---------------- model constants ----------------
#define Bv   2
#define Tv   2048
#define Dv   512
#define Hv   8
#define DHv  64
#define Lv   8
#define FFv  2048
#define Vv   32000
#define Mv   (Bv * Tv)        // 4096 token rows

#if __has_builtin(__builtin_amdgcn_tensor_load_to_lds)
#define HAVE_TDM 1
#else
#define HAVE_TDM 0
#endif

// ---------------- vector types ----------------
typedef __attribute__((ext_vector_type(16))) __bf16 v16bf;
typedef __attribute__((ext_vector_type(8)))  __bf16 v8bf;
typedef __attribute__((ext_vector_type(8)))  float  v8f;
typedef __attribute__((ext_vector_type(4)))  unsigned int u32x4;
typedef __attribute__((ext_vector_type(8)))  int i32x8;
typedef __attribute__((ext_vector_type(4)))  int i32x4;

union F8 { v8f v; float f[8]; };

// D = A(16x32 bf16) * B(32x16 bf16) + C(16x16 f32)
__device__ static inline v8f wmma_bf16(v16bf a, v16bf b, v8f c) {
    return __builtin_amdgcn_wmma_f32_16x16x32_bf16(
        false, a, false, b, (short)0, c, false, false);
}

// A-operand (16x32, 16-bit), k order chosen here for A and B alike: lane (m = lane&15) holds
//   K = (lane>>4)*16 + i
// -> 16 contiguous halves at p, loaded as two 16B segments at p and p+8.
__device__ static inline v16bf load_a_frag(const __bf16* p) {
    v8bf lo = *(const v8bf*)(p);
    v8bf hi = *(const v8bf*)(p + 8);
    return __builtin_shufflevector(lo, hi,
        0, 1, 2, 3, 4, 5, 6, 7, 8, 9, 10, 11, 12, 13, 14, 15);
}
// B-operand: 16 contiguous halves, loaded as two 16B pieces (80B LDS pitch
// is only 16B aligned).
__device__ static inline v16bf load_b_frag(const __bf16* p) {
    v8bf lo = *(const v8bf*)(p);
    v8bf hi = *(const v8bf*)(p + 8);
    return __builtin_shufflevector(lo, hi,
        0, 1, 2, 3, 4, 5, 6, 7, 8, 9, 10, 11, 12, 13, 14, 15);
}

#if HAVE_TDM
// ---- Tensor Data Mover: 2-D tile (tile_d1 rows x tile_d0 elems, 2B elems)
// global row pitch = stride0 elems; LDS gets 64B rows padded by 16B (pitch 80B).
// Descriptor packing per CDNA5 ISA.
__device__ static inline void tdm_load_2d(unsigned lds_addr, const void* gptr,
                                          unsigned tensor_d0, unsigned tensor_d1,
                                          unsigned long long stride0,
                                          unsigned tile_d0, unsigned tile_d1) {
    const unsigned long long ga = (unsigned long long)(uintptr_t)gptr;
    u32x4 g0;
    g0[0] = 1u;                                   // count=1, user descriptor
    g0[1] = lds_addr;                             // lds_addr [63:32]
    g0[2] = (unsigned)ga;                         // global_addr [95:64]
    g0[3] = (unsigned)(ga >> 32) | (2u << 30);    // global_addr hi | type=2
    i32x8 g1;
    // data_size=1(2B) | pad_enable | pad_interval=3(16 dw=64B) | pad_amount=3(4 dw=16B)
    g1[0] = (int)((1u << 16) | (1u << 20) | (3u << 22) | (3u << 25));
    g1[1] = (int)((tensor_d0 & 0xFFFFu) << 16);                     // dim0 lo
    g1[2] = (int)((tensor_d0 >> 16) | ((tensor_d1 & 0xFFFFu) << 16));
    g1[3] = (int)((tensor_d1 >> 16) | (tile_d0 << 16));
    g1[4] = (int)(tile_d1 & 0xFFFFu);                               // tile_dim1, tile_dim2=0
    g1[5] = (int)(unsigned)stride0;                                 // dim0_stride lo
    g1[6] = (int)(unsigned)((stride0 >> 32) & 0xFFFFu);             // hi | dim1_stride=0
    g1[7] = 0;
    const i32x4 z4 = {0, 0, 0, 0};
    const i32x8 z8 = {0, 0, 0, 0, 0, 0, 0, 0};
    __builtin_amdgcn_tensor_load_to_lds(g0, g1, z4, z4, z8, 0);
}
#endif

// ---------------- weight transpose + fp32->bf16 convert ----------------
__global__ __launch_bounds__(256) void transpose_to_bf16(
    const float* __restrict__ in, __bf16* __restrict__ out, int rows, int cols) {
    const size_t slice = (size_t)rows * cols;
    const float* ip = in + slice * blockIdx.z;
    __bf16* op = out + slice * blockIdx.z;
    const int idx = blockIdx.x * 256 + threadIdx.x;
    if (idx < (int)slice) {
        const int r = idx / cols, c = idx % cols;
        op[(size_t)c * rows + r] = (__bf16)ip[idx];
    }
}

// ---------------- embedding + sinusoidal positional encoding ----------------
__global__ __launch_bounds__(256) void embed_kernel(
    const int* __restrict__ x, const float* __restrict__ emb,
    float* __restrict__ h) {
    const int row = blockIdx.x;
    const int t   = row & (Tv - 1);
    const int tok = x[row];
    const float* e = emb + (size_t)tok * Dv;
    float* hp = h + (size_t)row * Dv;
    for (int d = threadIdx.x; d < Dv; d += 256) {
        const float freq = __expf((float)(d & ~1) * (-9.210340371976184f / (float)Dv));
        const float ang  = (float)t * freq;
        const float pe   = (d & 1) ? __cosf(ang) : __sinf(ang);
        hp[d] = e[d] + pe;
    }
}

// ---------------- LayerNorm (fp32 in -> bf16 out) ----------------
__global__ __launch_bounds__(128) void ln_kernel(
    const float* __restrict__ h, const float* __restrict__ scale,
    const float* __restrict__ bias, __bf16* __restrict__ y) {
    const int row = blockIdx.x;
    const float* xp = h + (size_t)row * Dv;
    float loc[4];
    float s = 0.f;
    #pragma unroll
    for (int i = 0; i < 4; ++i) { loc[i] = xp[threadIdx.x + i * 128]; s += loc[i]; }
    __shared__ float red[128];
    red[threadIdx.x] = s;
    __syncthreads();
    for (int st = 64; st > 0; st >>= 1) {
        if (threadIdx.x < st) red[threadIdx.x] += red[threadIdx.x + st];
        __syncthreads();
    }
    const float mu = red[0] * (1.f / (float)Dv);
    __syncthreads();
    float vs = 0.f;
    #pragma unroll
    for (int i = 0; i < 4; ++i) { const float d = loc[i] - mu; vs += d * d; }
    red[threadIdx.x] = vs;
    __syncthreads();
    for (int st = 64; st > 0; st >>= 1) {
        if (threadIdx.x < st) red[threadIdx.x] += red[threadIdx.x + st];
        __syncthreads();
    }
    const float rstd = rsqrtf(red[0] * (1.f / (float)Dv) + 1e-5f);
    #pragma unroll
    for (int i = 0; i < 4; ++i) {
        const int c = threadIdx.x + i * 128;
        y[(size_t)row * Dv + c] = (__bf16)((loc[i] - mu) * rstd * scale[c] + bias[c]);
    }
}

// ---------------- fused bf16 WMMA GEMM: out = A[MxK] * Bt[NxK]^T + bias ----
// 128x128 block tile, 8 waves of 32x64. TDM double-buffers 128x32 A/B panels
// into LDS (80B padded pitch); wave 0 drives the DMA + TENSORcnt.
// MODE 0: QKV scatter (Q scaled 1/sqrt(DH); V stored transposed [bh][d][t])
// MODE 1: residual add into fp32 h   MODE 2: exact GELU -> bf16
// MODE 3: plain fp32 store (logits)
#define LDS_PITCH 40   // halves per row (32 data + 8 pad)
template <int MODE>
__global__ __launch_bounds__(256) void gemm_bf16_kernel(
    const __bf16* __restrict__ A, const __bf16* __restrict__ Bt,
    const float* __restrict__ bias, int M, int N, int K,
    float* __restrict__ outf, __bf16* __restrict__ outb,
    __bf16* __restrict__ Qo, __bf16* __restrict__ Ko, __bf16* __restrict__ Vto) {
    const int lane = threadIdx.x & 31;
    const int wave = threadIdx.x >> 5;
    const int lh = lane >> 4, lm = lane & 15;
    const int waveM = (wave & 3) * 32;           // 4 waves along M
    const int waveN = (wave >> 2) * 64;          // 2 waves along N
    const int rowBase = blockIdx.y * 128 + waveM;
    const int colBase = blockIdx.x * 128 + waveN;

    F8 acc[2][4];
    #pragma unroll
    for (int rt = 0; rt < 2; ++rt)
        #pragma unroll
        for (int ct = 0; ct < 4; ++ct)
            #pragma unroll
            for (int r = 0; r < 8; ++r) acc[rt][ct].f[r] = 0.f;

    const int nk = K >> 5;

#if HAVE_TDM
    __shared__ __align__(16) __bf16 Asm[2][128 * LDS_PITCH];
    __shared__ __align__(16) __bf16 Bsm[2][128 * LDS_PITCH];
    const __bf16* gA = A + (size_t)(blockIdx.y * 128) * K;
    const __bf16* gB = Bt + (size_t)(blockIdx.x * 128) * K;
    if (wave == 0) {
        tdm_load_2d((unsigned)(uintptr_t)&Asm[0][0], gA, (unsigned)K, 1u << 20,
                    (unsigned long long)K, 32u, 128u);
        tdm_load_2d((unsigned)(uintptr_t)&Bsm[0][0], gB, (unsigned)K, 1u << 20,
                    (unsigned long long)K, 32u, 128u);
    }
    for (int kc = 0; kc < nk; ++kc) {
        const int buf = kc & 1;
        if (wave == 0) {
            if (kc + 1 < nk) {
                tdm_load_2d((unsigned)(uintptr_t)&Asm[buf ^ 1][0], gA + (kc + 1) * 32,
                            (unsigned)K, 1u << 20, (unsigned long long)K, 32u, 128u);
                tdm_load_2d((unsigned)(uintptr_t)&Bsm[buf ^ 1][0], gB + (kc + 1) * 32,
                            (unsigned)K, 1u << 20, (unsigned long long)K, 32u, 128u);
                __builtin_amdgcn_s_wait_tensorcnt(2);   // stage kc complete (in-order)
            } else {
                __builtin_amdgcn_s_wait_tensorcnt(0);
            }
        }
        __syncthreads();                                // panels ready for all waves
        v16bf a[2];
        #pragma unroll
        for (int rt = 0; rt < 2; ++rt)
            a[rt] = load_a_frag(&Asm[buf][(waveM + rt * 16 + lm) * LDS_PITCH + lh * 16]);
        #pragma unroll
        for (int ct = 0; ct < 4; ++ct) {
            const v16bf b =
                load_b_frag(&Bsm[buf][(waveN + ct * 16 + lm) * LDS_PITCH + lh * 16]);
            #pragma unroll
            for (int rt = 0; rt < 2; ++rt)
                acc[rt][ct].v = wmma_bf16(a[rt], b, acc[rt][ct].v);
        }
        __syncthreads();                                // done reading before overwrite
    }
#else
    for (int kc = 0; kc < nk; ++kc) {
        v16bf a[2];
        #pragma unroll
        for (int rt = 0; rt < 2; ++rt) {
            const __bf16* pa = A + (size_t)(rowBase + rt * 16 + lm) * K + (kc << 5) + lh * 16;
            if (kc + 2 < nk) __builtin_prefetch(pa + 64, 0, 0);
            a[rt] = load_a_frag(pa);
        }
        #pragma unroll
        for (int ct = 0; ct < 4; ++ct) {
            const __bf16* pb = Bt + (size_t)(colBase + ct * 16 + lm) * K + (kc << 5) + lh * 16;
            const v16bf b = *(const v16bf*)pb;
            #pragma unroll
            for (int rt = 0; rt < 2; ++rt)
                acc[rt][ct].v = wmma_bf16(a[rt], b, acc[rt][ct].v);
        }
    }
#endif

    // epilogue: C layout -> elem r: row = base + (lane>>4)*8 + r, col = base + (lane&15)
    #pragma unroll
    for (int rt = 0; rt < 2; ++rt)
        #pragma unroll
        for (int ct = 0; ct < 4; ++ct)
            #pragma unroll
            for (int r = 0; r < 8; ++r) {
                const int row = rowBase + rt * 16 + lh * 8 + r;
                const int c   = colBase + ct * 16 + lm;
                const float v = acc[rt][ct].f[r] + bias[c];
                if (MODE == 0) {
                    const int bb  = row >> 11;
                    const int tt  = row & (Tv - 1);
                    const int sel = c >> 9;
                    const int wc  = c & (Dv - 1);
                    const int hd  = wc >> 6;
                    const int dd  = wc & 63;
                    if (sel == 0)
                        Qo[(((size_t)bb * Hv + hd) * Tv + tt) * DHv + dd] = (__bf16)(v * 0.125f);
                    else if (sel == 1)
                        Ko[(((size_t)bb * Hv + hd) * Tv + tt) * DHv + dd] = (__bf16)v;
                    else
                        Vto[(((size_t)bb * Hv + hd) * DHv + dd) * Tv + tt] = (__bf16)v;
                } else if (MODE == 1) {
                    const size_t idx = (size_t)row * N + c;
                    outf[idx] = outf[idx] + v;
                } else if (MODE == 2) {
                    outb[(size_t)row * N + c] =
                        (__bf16)(0.5f * v * (1.0f + erff(v * 0.70710678118f)));
                } else {
                    outf[(size_t)row * N + c] = v;
                }
            }
}

// ---------------- flash attention: 1 wave = 16 query rows ----------------
__global__ __launch_bounds__(32) void attn_kernel(
    const __bf16* __restrict__ Q, const __bf16* __restrict__ Kb,
    const __bf16* __restrict__ Vt, __bf16* __restrict__ O) {
    const int bh   = blockIdx.y;
    const int t0   = blockIdx.x * 16;
    const int lane = threadIdx.x & 31;
    const int lh = lane >> 4, lm = lane & 15;
    __shared__ __align__(32) __bf16 Pl[16 * 32];

    v16bf qa[2];
    #pragma unroll
    for (int kc = 0; kc < 2; ++kc) {
        const __bf16* pq = Q + ((size_t)bh * Tv + (t0 + lm)) * DHv + kc * 32 + lh * 16;
        qa[kc] = load_a_frag(pq);
    }

    F8 o[4];
    #pragma unroll
    for (int ot = 0; ot < 4; ++ot)
        #pragma unroll
        for (int r = 0; r < 8; ++r) o[ot].f[r] = 0.f;
    float mi[8], li[8];
    #pragma unroll
    for (int r = 0; r < 8; ++r) { mi[r] = -1e30f; li[r] = 0.f; }

    const int kbmax = (t0 + 15) >> 5;
    for (int kb = 0; kb <= kbmax; ++kb) {
        const int ks = kb * 32;
        F8 s[2];
        #pragma unroll
        for (int ct = 0; ct < 2; ++ct) {
            #pragma unroll
            for (int r = 0; r < 8; ++r) s[ct].f[r] = 0.f;
            #pragma unroll
            for (int kc = 0; kc < 2; ++kc) {
                const __bf16* pk =
                    Kb + ((size_t)bh * Tv + (ks + ct * 16 + lm)) * DHv + kc * 32 + lh * 16;
                const v16bf kf = *(const v16bf*)pk;
                s[ct].v = wmma_bf16(qa[kc], kf, s[ct].v);
            }
        }
        float p[2][8];
        #pragma unroll
        for (int r = 0; r < 8; ++r) {
            const int t = t0 + lh * 8 + r;
            const float v0 = (ks + lm <= t)      ? s[0].f[r] : -1e30f;
            const float v1 = (ks + 16 + lm <= t) ? s[1].f[r] : -1e30f;
            float mx = fmaxf(v0, v1);
            #pragma unroll
            for (int d = 1; d < 16; d <<= 1) mx = fmaxf(mx, __shfl_xor(mx, d, 16));
            const float mnew  = fmaxf(mi[r], mx);
            const float alpha = __expf(mi[r] - mnew);
            const float p0 = __expf(v0 - mnew);
            const float p1 = __expf(v1 - mnew);
            float rs = p0 + p1;
            #pragma unroll
            for (int d = 1; d < 16; d <<= 1) rs += __shfl_xor(rs, d, 16);
            li[r] = li[r] * alpha + rs;
            mi[r] = mnew;
            #pragma unroll
            for (int ot = 0; ot < 4; ++ot) o[ot].f[r] *= alpha;
            p[0][r] = p0; p[1][r] = p1;
        }
        #pragma unroll
        for (int ct = 0; ct < 2; ++ct)
            #pragma unroll
            for (int r = 0; r < 8; ++r)
                Pl[(lh * 8 + r) * 32 + ct * 16 + lm] = (__bf16)p[ct][r];
        asm volatile("s_wait_dscnt 0" ::: "memory");
        const v16bf pf = load_a_frag(&Pl[lm * 32 + lh * 16]);
        #pragma unroll
        for (int ot = 0; ot < 4; ++ot) {
            const __bf16* pv = Vt + ((size_t)bh * DHv + ot * 16 + lm) * Tv + ks + lh * 16;
            const v16bf vf = *(const v16bf*)pv;
            o[ot].v = wmma_bf16(pf, vf, o[ot].v);
        }
        asm volatile("s_wait_dscnt 0" ::: "memory");
    }

    const int b = bh >> 3, hh = bh & 7;
    #pragma unroll
    for (int r = 0; r < 8; ++r) {
        const int t = t0 + lh * 8 + r;
        const float inv = 1.f / li[r];
        #pragma unroll
        for (int ot = 0; ot < 4; ++ot)
            O[((size_t)(b * Tv + t)) * Dv + hh * DHv + ot * 16 + lm] =
                (__bf16)(o[ot].f[r] * inv);
    }
}

// ---------------- host orchestration ----------------
extern "C" void kernel_launch(void* const* d_in, const int* in_sizes, int n_in,
                              void* d_out, int out_size, void* d_ws, size_t ws_size,
                              hipStream_t stream) {
    (void)in_sizes; (void)n_in; (void)out_size; (void)ws_size;
    const int*   x     = (const int*)  d_in[0];
    const float* embw  = (const float*)d_in[1];
    const float* ln1_s = (const float*)d_in[2];
    const float* ln1_b = (const float*)d_in[3];
    const float* qkv_w = (const float*)d_in[4];
    const float* qkv_b = (const float*)d_in[5];
    const float* afc_w = (const float*)d_in[6];
    const float* afc_b = (const float*)d_in[7];
    const float* ln2_s = (const float*)d_in[8];
    const float* ln2_b = (const float*)d_in[9];
    const float* ff_w1 = (const float*)d_in[10];
    const float* ff_b1 = (const float*)d_in[11];
    const float* ff_w2 = (const float*)d_in[12];
    const float* ff_b2 = (const float*)d_in[13];
    const float* lnf_s = (const float*)d_in[14];
    const float* lnf_b = (const float*)d_in[15];
    const float* fc_w  = (const float*)d_in[16];
    const float* fc_b  = (const float*)d_in[17];

    char* ws = (char*)d_ws;
    size_t off = 0;
    auto alloc = [&](size_t bytes) -> void* {
        void* p = ws + off;
        off += (bytes + 255) & ~(size_t)255;
        return p;
    };

    __bf16* qkvT = (__bf16*)alloc((size_t)Lv * 3 * Dv * Dv * 2);
    __bf16* afcT = (__bf16*)alloc((size_t)Lv * Dv * Dv * 2);
    __bf16* w1T  = (__bf16*)alloc((size_t)Lv * FFv * Dv * 2);
    __bf16* w2T  = (__bf16*)alloc((size_t)Lv * Dv * FFv * 2);
    __bf16* fcT  = (__bf16*)alloc((size_t)Vv * Dv * 2);
    float*  h    = (float*) alloc((size_t)Mv * Dv * 4);
    __bf16* ybf  = (__bf16*)alloc((size_t)Mv * Dv * 2);
    __bf16* Qbf  = (__bf16*)alloc((size_t)Mv * Dv * 2);
    __bf16* Kbf  = (__bf16*)alloc((size_t)Mv * Dv * 2);
    __bf16* Vt   = (__bf16*)alloc((size_t)Mv * Dv * 2);
    __bf16* Obf  = (__bf16*)alloc((size_t)Mv * Dv * 2);
    __bf16* a1   = (__bf16*)alloc((size_t)Mv * FFv * 2);

    transpose_to_bf16<<<dim3((Dv * 3 * Dv + 255) / 256, 1, Lv), 256, 0, stream>>>(
        qkv_w, qkvT, Dv, 3 * Dv);
    transpose_to_bf16<<<dim3((Dv * Dv + 255) / 256, 1, Lv), 256, 0, stream>>>(
        afc_w, afcT, Dv, Dv);
    transpose_to_bf16<<<dim3((Dv * FFv + 255) / 256, 1, Lv), 256, 0, stream>>>(
        ff_w1, w1T, Dv, FFv);
    transpose_to_bf16<<<dim3((FFv * Dv + 255) / 256, 1, Lv), 256, 0, stream>>>(
        ff_w2, w2T, FFv, Dv);
    transpose_to_bf16<<<dim3((Dv * Vv + 255) / 256, 1, 1), 256, 0, stream>>>(
        fc_w, fcT, Dv, Vv);

    embed_kernel<<<Mv, 256, 0, stream>>>(x, embw, h);

    for (int l = 0; l < Lv; ++l) {
        ln_kernel<<<Mv, 128, 0, stream>>>(h, ln1_s + l * Dv, ln1_b + l * Dv, ybf);
        gemm_bf16_kernel<0><<<dim3(3 * Dv / 128, Mv / 128), 256, 0, stream>>>(
            ybf, qkvT + (size_t)l * 3 * Dv * Dv, qkv_b + l * 3 * Dv,
            Mv, 3 * Dv, Dv, nullptr, nullptr, Qbf, Kbf, Vt);
        attn_kernel<<<dim3(Tv / 16, Bv * Hv), 32, 0, stream>>>(Qbf, Kbf, Vt, Obf);
        gemm_bf16_kernel<1><<<dim3(Dv / 128, Mv / 128), 256, 0, stream>>>(
            Obf, afcT + (size_t)l * Dv * Dv, afc_b + l * Dv,
            Mv, Dv, Dv, h, nullptr, nullptr, nullptr, nullptr);
        ln_kernel<<<Mv, 128, 0, stream>>>(h, ln2_s + l * Dv, ln2_b + l * Dv, ybf);
        gemm_bf16_kernel<2><<<dim3(FFv / 128, Mv / 128), 256, 0, stream>>>(
            ybf, w1T + (size_t)l * FFv * Dv, ff_b1 + l * FFv,
            Mv, FFv, Dv, nullptr, a1, nullptr, nullptr, nullptr);
        gemm_bf16_kernel<1><<<dim3(Dv / 128, Mv / 128), 256, 0, stream>>>(
            a1, w2T + (size_t)l * Dv * FFv, ff_b2 + l * Dv,
            Mv, Dv, FFv, h, nullptr, nullptr, nullptr, nullptr);
    }

    ln_kernel<<<Mv, 128, 0, stream>>>(h, lnf_s, lnf_b, ybf);
    gemm_bf16_kernel<3><<<dim3(Vv / 128, Mv / 128), 256, 0, stream>>>(
        ybf, fcT, fc_b, Mv, Vv, Dv, (float*)d_out, nullptr, nullptr, nullptr, nullptr);
}
